// SS3D_88467736363573
// MI455X (gfx1250) — compile-verified, matrix operand order checked by reading, not run on hardware
//
#include <hip/hip_runtime.h>
#include <math.h>

// Problem constants (from reference)
#define BB 4
#define LL 1024
#define DD 96
#define HH 32
#define WW 32
#define DI 192
#define CIc 32
#define NN 16
#define RR 6
#define KK 4
#define C6 (6*DD)     // 576
#define D3 (3*DD)     // 288
#define BL (BB*LL)    // 4096
#define LC (DI*WW)    // 6144
#define PJ (RR+2*NN)  // 38 logical columns of the x-projection
#define PJS 40        // padded storage stride (16B-aligned rows)

typedef __attribute__((ext_vector_type(16))) _Float16 v16h;
typedef __attribute__((ext_vector_type(8)))  float    v8f;
typedef __attribute__((ext_vector_type(4)))  float    f4;
typedef __attribute__((ext_vector_type(4)))  _Float16 h4;

#define LDST 48   // LDS row stride in halves (96B: 16B-aligned rows)

__device__ __constant__ float c_ct[8] = {1.f, 0.70710678f, 0.f, -0.70710678f,
                                         -1.f, -0.70710678f, 0.f, 0.70710678f};
__device__ __constant__ float c_st[8] = {0.f, 0.70710678f, 1.f, 0.70710678f,
                                         0.f, -0.70710678f, -1.f, -0.70710678f};

__device__ __forceinline__ float siluf(float x)     { return x / (1.f + __expf(-x)); }
__device__ __forceinline__ float softplusf(float x) { return x > 20.f ? x : log1pf(__expf(x)); }

// ---------------------------------------------------------------------------
// WMMA GEMM:  C[M,N] = act( A[M,K] * W[N,K]^T + bias[N] )
// REQUIREMENTS (hold at all call sites):
//   M % 64 == 0; in the K%32==0 fast path lda, ldw are multiples of 4 and
//   A, W are 16B-aligned, so float4 loads are legal.
// 128 threads = 4 waves. Block tile 64x32: one shared A fragment + two B
// fragments per wave -> two v_wmma_f32_16x16x32_f16 per k-chunk.
// Staging is two-phase: float4 global gathers into registers, then packed
// f16 ds_store_b64 writes -> batched global_load_b128, single wait.
// act: 0=none, 1=silu, 2=softplus
// ---------------------------------------------------------------------------
__global__ void __launch_bounds__(128)
k_gemm(const float* __restrict__ A, const float* __restrict__ Wt,
       const float* __restrict__ bias, float* __restrict__ Cmat,
       int M, int N, int Kd, int lda, int ldw, int ldc, int act)
{
    __shared__ _Float16 Ash[64][LDST];
    __shared__ _Float16 Wsh[32][LDST];
    const int tid  = threadIdx.x;
    const int wid  = tid >> 5;
    const int lane = tid & 31;
    const int m0   = blockIdx.x * 64;
    const int n0   = blockIdx.y * 32;

    v8f acc0 = {}, acc1 = {};
    for (int kc = 0; kc < Kd; kc += 32) {
        if (kc + 32 <= Kd) {
            // ---- fast path: vectorized float4 staging, no K guard ----
            f4 av[4];
#pragma unroll
            for (int e = 0; e < 4; ++e) {               // A: 64x32 = 512 float4s
                int idx = tid + e * 128;
                int r = idx >> 3, q = idx & 7;
                av[e] = *(const f4*)&A[(size_t)(m0 + r) * lda + kc + q * 4];
            }
            f4 wv[2];
#pragma unroll
            for (int e = 0; e < 2; ++e) {               // W: 32x32 = 256 float4s
                int idx = tid + e * 128;
                int n = idx >> 3, q = idx & 7;
                int row = n0 + n;
                int rc  = row < N ? row : 0;            // clamp, branch-free
                f4 v = *(const f4*)&Wt[(size_t)rc * ldw + kc + q * 4];
                if (row >= N) v = (f4){0.f, 0.f, 0.f, 0.f};
                wv[e] = v;
            }
#pragma unroll
            for (int e = 0; e < 4; ++e) {
                int idx = tid + e * 128;
                int r = idx >> 3, q = idx & 7;
                h4 hv = {(_Float16)av[e][0], (_Float16)av[e][1],
                         (_Float16)av[e][2], (_Float16)av[e][3]};
                *(h4*)&Ash[r][q * 4] = hv;
            }
#pragma unroll
            for (int e = 0; e < 2; ++e) {
                int idx = tid + e * 128;
                int n = idx >> 3, q = idx & 7;
                h4 hv = {(_Float16)wv[e][0], (_Float16)wv[e][1],
                         (_Float16)wv[e][2], (_Float16)wv[e][3]};
                *(h4*)&Wsh[n][q * 4] = hv;
            }
        } else {
            // ---- K tail (only the dt projection: K=6): scalar, branch-free ----
#pragma unroll
            for (int e = 0; e < 16; ++e) {
                int idx = tid + e * 128;
                int r = idx >> 5, kk = idx & 31;
                int col = kc + kk;
                int cc  = col < Kd ? col : 0;
                float v = A[(size_t)(m0 + r) * lda + cc];
                Ash[r][kk] = (_Float16)(col < Kd ? v : 0.f);
            }
#pragma unroll
            for (int e = 0; e < 8; ++e) {
                int idx = tid + e * 128;
                int n = idx >> 5, kk = idx & 31;
                int row = n0 + n, col = kc + kk;
                bool ok = (row < N) && (col < Kd);
                int rc = row < N ? row : 0;
                int cc = col < Kd ? col : 0;
                float v = Wt[(size_t)rc * ldw + cc];
                Wsh[n][kk] = (_Float16)(ok ? v : 0.f);
            }
        }
        __syncthreads();

        v16h af, bf0, bf1;
        const int mrow = (wid << 4) + (lane & 15);
        const int ncol = lane & 15;
#pragma unroll
        for (int h = 0; h < 16; ++h) {
            int k = ((h & 8) << 1) + ((lane & 16) >> 1) + (h & 7);
            af[h]  = Ash[mrow][k];
            bf0[h] = Wsh[ncol][k];
            bf1[h] = Wsh[16 + ncol][k];
        }
        acc0 = __builtin_amdgcn_wmma_f32_16x16x32_f16(false, af, false, bf0,
                                                      (short)0, acc0, false, false);
        acc1 = __builtin_amdgcn_wmma_f32_16x16x32_f16(false, af, false, bf1,
                                                      (short)0, acc1, false, false);
        __syncthreads();
    }

    const int nA    = n0 + (lane & 15);
    const int nB    = n0 + 16 + (lane & 15);
    const int mbase = m0 + (wid << 4) + ((lane & 16) >> 1);
    float bv0 = (bias != nullptr && nA < N) ? bias[nA] : 0.f;
    float bv1 = (bias != nullptr && nB < N) ? bias[nB] : 0.f;
#pragma unroll
    for (int r = 0; r < 8; ++r) {
        int m = mbase + r;                              // always < M (M % 64 == 0)
        if (nA < N) {
            float v = acc0[r] + bv0;
            if (act == 1)      v = siluf(v);
            else if (act == 2) v = softplusf(v);
            Cmat[(size_t)m * ldc + nA] = v;
        }
        if (nB < N) {
            float v = acc1[r] + bv1;
            if (act == 1)      v = siluf(v);
            else if (act == 2) v = softplusf(v);
            Cmat[(size_t)m * ldc + nB] = v;
        }
    }
}

// ---------------------------------------------------------------------------
// Depthwise 3x3 conv, pad=1, on (B, L=H*W, C) row-major data.
// ---------------------------------------------------------------------------
__global__ void k_dw3x3(const float* __restrict__ in, const float* __restrict__ wgt,
                        const float* __restrict__ bias, float* __restrict__ out, int C)
{
    int idx = blockIdx.x * blockDim.x + threadIdx.x;
    int total = BB * LL * C;
    if (idx >= total) return;
    int c = idx % C;
    int l = (idx / C) % LL;
    int b = idx / (C * LL);
    int h = l / WW, w = l % WW;
    float acc = bias ? bias[c] : 0.f;
#pragma unroll
    for (int dy = 0; dy < 3; ++dy) {
        int hh = h + dy - 1;
        if (hh < 0 || hh >= HH) continue;
#pragma unroll
        for (int dx = 0; dx < 3; ++dx) {
            int ww2 = w + dx - 1;
            if (ww2 < 0 || ww2 >= WW) continue;
            acc += wgt[c * 9 + dy * 3 + dx] *
                   in[((size_t)b * LL + hh * WW + ww2) * C + c];
        }
    }
    out[idx] = acc;
}

// ---------------------------------------------------------------------------
// 8x8 block spectral filter: y = irfft2(rfft2(p) * w), per (b, channel, block).
// Twiddles from constant tables (8-point DFT).
// ---------------------------------------------------------------------------
__global__ void k_fftfilt(const float* __restrict__ f0, const float* __restrict__ fw,
                          float* __restrict__ f1)
{
    int t = blockIdx.x * blockDim.x + threadIdx.x;
    if (t >= BB * C6 * 16) return;
    int blk = t % 16;
    int c   = (t / 16) % C6;
    int b   = t / (16 * C6);
    int br = blk >> 2, bc = blk & 3;

    float p[8][8];
    for (int m = 0; m < 8; ++m)
        for (int n = 0; n < 8; ++n)
            p[m][n] = f0[((size_t)b * LL + (br * 8 + m) * WW + bc * 8 + n) * C6 + c];

    float PR[8][5], PI[8][5];
    for (int u = 0; u < 8; ++u)
        for (int v = 0; v < 5; ++v) {
            float cr = 0.f, ci = 0.f;
            for (int m = 0; m < 8; ++m)
                for (int n = 0; n < 8; ++n) {
                    int a = (u * m + v * n) & 7;
                    cr += p[m][n] * c_ct[a];
                    ci -= p[m][n] * c_st[a];
                }
            float wv = fw[(size_t)c * 40 + u * 5 + v];
            PR[u][v] = cr * wv;
            PI[u][v] = ci * wv;
        }

    for (int m = 0; m < 8; ++m)
        for (int n = 0; n < 8; ++n) {
            float acc = 0.f;
            for (int u = 0; u < 8; ++u)
                for (int v = 0; v < 8; ++v) {
                    float Yr, Yi;
                    if (v <= 4) { Yr = PR[u][v]; Yi = PI[u][v]; }
                    else {
                        int u2 = (8 - u) & 7;
                        int v2 = 8 - v;
                        Yr = PR[u2][v2]; Yi = -PI[u2][v2];
                    }
                    int a = (u * m + v * n) & 7;
                    acc += Yr * c_ct[a] - Yi * c_st[a];
                }
            f1[((size_t)b * LL + (br * 8 + m) * WW + bc * 8 + n) * C6 + c] = acc * (1.f / 64.f);
        }
}

// gelu(x1) * x2 gate across channel split
__global__ void k_gelugate(const float* __restrict__ f2, float* __restrict__ g)
{
    int t = blockIdx.x * blockDim.x + threadIdx.x;
    if (t >= BL * D3) return;
    int r = t / D3, c = t % D3;
    float a  = f2[(size_t)r * C6 + c];
    float x2 = f2[(size_t)r * C6 + D3 + c];
    float ge = 0.5f * a * (1.f + erff(a * 0.70710678f));
    g[t] = ge * x2;
}

// causal depthwise conv1d (K=4) + SiLU for spatial scans; reversal folded into gather
__global__ void k_conv1d_sp(const float* __restrict__ xz, const float* __restrict__ cw,
                            const float* __restrict__ cb, float* __restrict__ u)
{
    int idx = blockIdx.x * blockDim.x + threadIdx.x;
    if (idx >= 2 * BB * LL * DI) return;
    int c   = idx % DI;
    int l   = (idx / DI) % LL;
    int b   = (idx / (DI * LL)) % BB;
    int dir = idx / (DI * LL * BB);
    float acc = cb[dir * DI + c];
#pragma unroll
    for (int k = 0; k < KK; ++k) {
        int lp = l - (KK - 1) + k;
        if (lp < 0) continue;
        int ls = dir ? (LL - 1 - lp) : lp;
        acc += cw[(dir * DI + c) * KK + k] *
               xz[((size_t)b * LL + ls) * (2 * DI) + c];
    }
    u[idx] = siluf(acc);
}

// causal conv1d + SiLU for channel scans; dirs 0,1 = H-scan, 2,3 = W-scan
__global__ void k_conv1d_ch(const float* __restrict__ xz, const float* __restrict__ cw,
                            const float* __restrict__ cb, float* __restrict__ u)
{
    int idx = blockIdx.x * blockDim.x + threadIdx.x;
    if (idx >= 4 * BB * LC * CIc) return;
    int c   = idx % CIc;
    int j   = (idx / CIc) % LC;
    int b   = (idx / (CIc * LC)) % BB;
    int dir = idx / (CIc * LC * BB);
    float acc = cb[dir * CIc + c];
#pragma unroll
    for (int k = 0; k < KK; ++k) {
        int jp = j - (KK - 1) + k;
        if (jp < 0) continue;
        int js = (dir & 1) ? (LC - 1 - jp) : jp;
        float val;
        if (dir < 2) {
            int d = js / WW, w2 = js % WW;                    // c is h
            val = xz[((size_t)b * LL + c * WW + w2) * (2 * DI) + d];
        } else {
            int d = js / HH, h2 = js % HH;                    // c is w
            val = xz[((size_t)b * LL + h2 * WW + c) * (2 * DI) + d];
        }
        acc += cw[(dir * CIc + c) * KK + k] * val;
    }
    u[idx] = siluf(acc);
}

// spatial selective scan: block = (dir, b), thread = channel; z-gated; writes at orig l
__global__ void __launch_bounds__(DI)
k_scan_sp(const float* __restrict__ u, const float* __restrict__ proj,
          const float* __restrict__ delta, const float* __restrict__ xz,
          const float* __restrict__ A_log, const float* __restrict__ Dp,
          float* __restrict__ ys)
{
    int g = blockIdx.x;          // dir*BB + b
    int dir = g / BB, b = g % BB;
    int c = threadIdx.x;
    float An[NN];
#pragma unroll
    for (int n = 0; n < NN; ++n) An[n] = -__expf(A_log[((size_t)dir * DI + c) * NN + n]);
    float Dv = Dp[dir * DI + c];
    float s[NN];
#pragma unroll
    for (int n = 0; n < NN; ++n) s[n] = 0.f;

    for (int l = 0; l < LL; ++l) {
        size_t row = (size_t)g * LL + l;
        float d  = delta[row * DI + c];
        float uu = u[row * DI + c];
        if (l + 8 < LL) __builtin_prefetch(&u[(row + 8) * DI + c], 0, 3);
        float y = uu * Dv;
#pragma unroll
        for (int n = 0; n < NN; ++n) {
            float dA = __expf(d * An[n]);
            s[n] = dA * s[n] + d * proj[row * PJS + RR + n] * uu;
            y += proj[row * PJS + RR + NN + n] * s[n];
        }
        int ls = dir ? (LL - 1 - l) : l;
        float z = xz[((size_t)b * LL + ls) * (2 * DI) + DI + c];
        y *= siluf(z);
        ys[((size_t)g * LL + ls) * DI + c] = y;
    }
}

// channel selective scan (no gate); writes at original j
__global__ void __launch_bounds__(CIc)
k_scan_ch(const float* __restrict__ u, const float* __restrict__ proj,
          const float* __restrict__ delta, const float* __restrict__ A_log,
          const float* __restrict__ Dp, float* __restrict__ ych)
{
    int g = blockIdx.x;          // dir*BB + b
    int dir = g / BB;
    int c = threadIdx.x;
    float An[NN];
#pragma unroll
    for (int n = 0; n < NN; ++n) An[n] = -__expf(A_log[((size_t)dir * CIc + c) * NN + n]);
    float Dv = Dp[dir * CIc + c];
    float s[NN];
#pragma unroll
    for (int n = 0; n < NN; ++n) s[n] = 0.f;

    for (int j = 0; j < LC; ++j) {
        size_t row = (size_t)g * LC + j;
        float d  = delta[row * CIc + c];
        float uu = u[row * CIc + c];
        if (j + 8 < LC) __builtin_prefetch(&u[(row + 8) * CIc + c], 0, 3);
        float y = uu * Dv;
#pragma unroll
        for (int n = 0; n < NN; ++n) {
            float dA = __expf(d * An[n]);
            s[n] = dA * s[n] + d * proj[row * PJS + RR + n] * uu;
            y += proj[row * PJS + RR + NN + n] * s[n];
        }
        int js = (dir & 1) ? (LC - 1 - j) : j;
        ych[((size_t)g * LC + js) * CIc + c] = y;
    }
}

// total = ys_fwd + ys_rev + lr + freq + (h-scan pair) + (w-scan pair)
__global__ void k_combine(const float* __restrict__ ys, const float* __restrict__ ych,
                          const float* __restrict__ lr, const float* __restrict__ freq,
                          float* __restrict__ total)
{
    int t = blockIdx.x * blockDim.x + threadIdx.x;
    if (t >= BL * DI) return;
    int di = t % DI;
    int l  = (t / DI) % LL;
    int b  = t / (DI * LL);
    int h = l / WW, w = l % WW;
    size_t jh = (size_t)di * WW + w;
    size_t jw = (size_t)di * HH + h;
    float v = ys[((size_t)(0 * BB + b) * LL + l) * DI + di]
            + ys[((size_t)(1 * BB + b) * LL + l) * DI + di]
            + lr[t] + freq[t]
            + ych[(((size_t)(0 * BB + b)) * LC + jh) * CIc + h]
            + ych[(((size_t)(1 * BB + b)) * LC + jh) * CIc + h]
            + ych[(((size_t)(2 * BB + b)) * LC + jw) * CIc + w]
            + ych[(((size_t)(3 * BB + b)) * LC + jw) * CIc + w];
    total[t] = v;
}

// ---------------------------------------------------------------------------
extern "C" void kernel_launch(void* const* d_in, const int* in_sizes, int n_in,
                              void* d_out, int out_size, void* d_ws, size_t ws_size,
                              hipStream_t stream)
{
    (void)in_sizes; (void)n_in; (void)out_size; (void)ws_size;
    const float* hs         = (const float*)d_in[0];
    const float* lr1_dw_w   = (const float*)d_in[1];
    const float* lr1_dw_b   = (const float*)d_in[2];
    const float* lr1_pw_w   = (const float*)d_in[3];
    const float* lr1_pw_b   = (const float*)d_in[4];
    const float* lr2_dw_w   = (const float*)d_in[5];
    const float* lr2_dw_b   = (const float*)d_in[6];
    const float* lr2_pw_w   = (const float*)d_in[7];
    const float* lr2_pw_b   = (const float*)d_in[8];
    const float* in_proj_w  = (const float*)d_in[9];
    const float* sp_conv_w  = (const float*)d_in[10];
    const float* sp_conv_b  = (const float*)d_in[11];
    const float* sp_xproj_w = (const float*)d_in[12];
    const float* sp_dt_w    = (const float*)d_in[13];
    const float* sp_dt_b    = (const float*)d_in[14];
    const float* sp_A_log   = (const float*)d_in[15];
    const float* sp_Dv      = (const float*)d_in[16];
    const float* ch_conv_w  = (const float*)d_in[17];
    const float* ch_conv_b  = (const float*)d_in[18];
    const float* ch_xproj_w = (const float*)d_in[19];
    const float* ch_dt_w    = (const float*)d_in[20];
    const float* ch_dt_b    = (const float*)d_in[21];
    const float* ch_A_log   = (const float*)d_in[22];
    const float* ch_Dv      = (const float*)d_in[23];
    const float* out_proj_w = (const float*)d_in[24];
    const float* proj_in_w  = (const float*)d_in[25];
    const float* dwconv_w   = (const float*)d_in[26];
    const float* fft_w      = (const float*)d_in[27];
    const float* proj_out_w = (const float*)d_in[28];

    float* ws = (float*)d_ws;
    size_t off = 0;
    auto alloc = [&](size_t nelem) { float* p = ws + off; off += nelem; return p; };

    float* t1    = alloc((size_t)BL * DD);
    float* t2    = alloc((size_t)BL * DD);
    float* lrb   = alloc((size_t)BL * DI);
    float* f0    = alloc((size_t)BL * C6);
    float* f1    = alloc((size_t)BL * C6);
    float* gb    = alloc((size_t)BL * D3);
    float* freqb = alloc((size_t)BL * DI);
    float* xzb   = alloc((size_t)BL * 2 * DI);
    float* usp   = alloc((size_t)2 * BL * DI);
    float* psp   = alloc((size_t)2 * BL * PJS);
    float* dsp   = alloc((size_t)2 * BL * DI);
    float* ysp   = alloc((size_t)2 * BL * DI);
    float* uch   = alloc((size_t)4 * BB * LC * CIc);
    float* pch   = alloc((size_t)4 * BB * LC * PJS);
    float* dch   = alloc((size_t)4 * BB * LC * CIc);
    float* ychb  = alloc((size_t)4 * BB * LC * CIc);
    float* totb  = alloc((size_t)BL * DI);

    dim3 blk128(128);
    auto ggrid = [](int M, int N) { return dim3((unsigned)(M / 64), (unsigned)((N + 31) / 32)); };

    // ---- lr path ----
    int n1 = BL * DD;
    k_dw3x3<<<(n1 + 255) / 256, 256, 0, stream>>>(hs, lr1_dw_w, lr1_dw_b, t1, DD);
    k_gemm<<<ggrid(BL, DD), blk128, 0, stream>>>(t1, lr1_pw_w, lr1_pw_b, t2, BL, DD, DD, DD, DD, DD, 1);
    k_dw3x3<<<(n1 + 255) / 256, 256, 0, stream>>>(t2, lr2_dw_w, lr2_dw_b, t1, DD);
    k_gemm<<<ggrid(BL, DI), blk128, 0, stream>>>(t1, lr2_pw_w, lr2_pw_b, lrb, BL, DI, DD, DD, DD, DI, 0);

    // ---- freq path ----
    k_gemm<<<ggrid(BL, C6), blk128, 0, stream>>>(hs, proj_in_w, nullptr, f0, BL, C6, DD, DD, DD, C6, 0);
    int nf = BB * C6 * 16;
    k_fftfilt<<<(nf + 127) / 128, 128, 0, stream>>>(f0, fft_w, f1);
    int n6 = BL * C6;
    k_dw3x3<<<(n6 + 255) / 256, 256, 0, stream>>>(f1, dwconv_w, nullptr, f0, C6);   // f2 reuses f0
    int ng = BL * D3;
    k_gelugate<<<(ng + 255) / 256, 256, 0, stream>>>(f0, gb);
    k_gemm<<<ggrid(BL, DI), blk128, 0, stream>>>(gb, proj_out_w, nullptr, freqb, BL, DI, D3, D3, D3, DI, 0);

    // ---- in_proj ----
    k_gemm<<<ggrid(BL, 2 * DI), blk128, 0, stream>>>(hs, in_proj_w, nullptr, xzb, BL, 2 * DI, DD, DD, DD, 2 * DI, 0);

    // ---- spatial SSM ----
    int nus = 2 * BL * DI;
    k_conv1d_sp<<<(nus + 255) / 256, 256, 0, stream>>>(xzb, sp_conv_w, sp_conv_b, usp);
    for (int d2 = 0; d2 < 2; ++d2) {
        k_gemm<<<ggrid(BL, PJ), blk128, 0, stream>>>(
            usp + (size_t)d2 * BL * DI, sp_xproj_w + (size_t)d2 * PJ * DI, nullptr,
            psp + (size_t)d2 * BL * PJS, BL, PJ, DI, DI, DI, PJS, 0);
        k_gemm<<<ggrid(BL, DI), blk128, 0, stream>>>(
            psp + (size_t)d2 * BL * PJS, sp_dt_w + (size_t)d2 * DI * RR, sp_dt_b + d2 * DI,
            dsp + (size_t)d2 * BL * DI, BL, DI, RR, PJS, RR, DI, 2);
    }
    k_scan_sp<<<2 * BB, DI, 0, stream>>>(usp, psp, dsp, xzb, sp_A_log, sp_Dv, ysp);

    // ---- channel SSM ----
    int nuc = 4 * BB * LC * CIc;
    k_conv1d_ch<<<(nuc + 255) / 256, 256, 0, stream>>>(xzb, ch_conv_w, ch_conv_b, uch);
    int MC = BB * LC;
    for (int d4 = 0; d4 < 4; ++d4) {
        k_gemm<<<ggrid(MC, PJ), blk128, 0, stream>>>(
            uch + (size_t)d4 * MC * CIc, ch_xproj_w + (size_t)d4 * PJ * CIc, nullptr,
            pch + (size_t)d4 * MC * PJS, MC, PJ, CIc, CIc, CIc, PJS, 0);
        k_gemm<<<ggrid(MC, CIc), blk128, 0, stream>>>(
            pch + (size_t)d4 * MC * PJS, ch_dt_w + (size_t)d4 * CIc * RR, ch_dt_b + d4 * CIc,
            dch + (size_t)d4 * MC * CIc, MC, CIc, RR, PJS, RR, CIc, 2);
    }
    k_scan_ch<<<4 * BB, CIc, 0, stream>>>(uch, pch, dch, ch_A_log, ch_Dv, ychb);

    // ---- combine + out_proj ----
    int nt = BL * DI;
    k_combine<<<(nt + 255) / 256, 256, 0, stream>>>(ysp, ychb, lrb, freqb, totb);
    k_gemm<<<ggrid(BL, DD), blk128, 0, stream>>>(totb, out_proj_w, nullptr, (float*)d_out,
                                                 BL, DD, DI, DI, DI, DD, 0);
}
